// SlotAttention_5935644803518
// MI455X (gfx1250) — matrix operand order repaired and checked by reading, not compiled, not run on hardware
//
#include <hip/hip_runtime.h>
#include <hip/hip_bf16.h>

typedef unsigned short u16;
typedef __attribute__((ext_vector_type(8)))  u16    v8u16;
typedef __attribute__((ext_vector_type(16))) u16    v16u16;
typedef __attribute__((ext_vector_type(16))) __bf16 v16bf;
typedef __attribute__((ext_vector_type(8)))  float  v8f;

#define LN_EPS 1e-5f
#define REN_EPS 1e-8f

static __device__ __forceinline__ u16 f2bf(float f) {
    unsigned u = __float_as_uint(f);
    unsigned r = u + 0x7FFFu + ((u >> 16) & 1u);   // round-to-nearest-even
    return (u16)(r >> 16);
}

static __device__ __forceinline__ v8f wmma_bf16(v16bf a, v16bf b, v8f c) {
    return __builtin_amdgcn_wmma_f32_16x16x32_bf16(
        /*neg_a=*/false, a, /*neg_b=*/false, b,
        /*c_mod=*/(short)0, c, /*reuse_a=*/false, /*reuse_b=*/false);
}

// A-matrix 16x32 bf16 fragment. row_ptr = row M (=lane&15) contiguous along K.
// lanes 0-15: K = kb..kb+15, lanes 16-31: K = kb+16..kb+31 (contiguous).
static __device__ __forceinline__ v16bf make_a_frag(const u16* row_ptr, int kb, int hi) {
    const int ko = hi * 16;
    v8u16 lo8 = *(const v8u16*)(row_ptr + kb + ko);
    v8u16 hi8 = *(const v8u16*)(row_ptr + kb + 8 + ko);
    v16u16 f;
#pragma unroll
    for (int i = 0; i < 8; ++i) { f[i] = lo8[i]; f[8 + i] = hi8[i]; }
    return __builtin_bit_cast(v16bf, f);
}

// B-matrix 32x16 bf16 fragment where memory holds B^T contiguous along K
// (i.e. row_ptr = output column N's data along K).
// lanes 0-15 hold K=kb..kb+15, lanes 16-31 hold K=kb+16..kb+31.
static __device__ __forceinline__ v16bf make_b_frag_rowk(const u16* row_ptr, int kb, int hi) {
    v16u16 f = *(const v16u16*)(row_ptr + kb + hi * 16);
    return __builtin_bit_cast(v16bf, f);
}

// B fragment from transposed LDS tile [d][j] (K contiguous per column).
// base = &lds[col * stride]; 16-byte aligned v8u16 pair.
static __device__ __forceinline__ v16bf make_b_frag_ldsT(const u16* base, int hi) {
    v8u16 a0 = *(const v8u16*)(base + hi * 16);
    v8u16 a1 = *(const v8u16*)(base + hi * 16 + 8);
    v16u16 f;
#pragma unroll
    for (int i = 0; i < 8; ++i) { f[i] = a0[i]; f[8 + i] = a1[i]; }
    return __builtin_bit_cast(v16bf, f);
}

// ---------------------------------------------------------------- small utils
__global__ void castw_kernel(const float* __restrict__ W, u16* __restrict__ Wb, int n) {
    int i = blockIdx.x * 256 + threadIdx.x;
    if (i < n) Wb[i] = f2bf(W[i]);
}

__global__ void zero_kernel(float* __restrict__ p, int n) {
    int i = blockIdx.x * 256 + threadIdx.x;
    if (i < n) p[i] = 0.f;
}

__global__ void init_slots_kernel(const float* __restrict__ noise, const float* __restrict__ mu,
                                  const float* __restrict__ lsig, float* __restrict__ slots) {
    int i = blockIdx.x * 256 + threadIdx.x;   // 65536
    int c = i & 255;
    slots[i] = mu[c] + __expf(lsig[c]) * noise[i];
}

__global__ void copy_kernel(const float* __restrict__ a, float* __restrict__ b, int n) {
    int i = blockIdx.x * 256 + threadIdx.x;
    if (i < n) b[i] = a[i];
}

__global__ void add_kernel(float* __restrict__ a, const float* __restrict__ b, int n) {
    int i = blockIdx.x * 256 + threadIdx.x;
    if (i < n) a[i] += b[i];
}

// Row LayerNorm (rows of length 256), one wave per row.
__global__ void ln_rows_kernel(const float* __restrict__ X, const float* __restrict__ g,
                               const float* __restrict__ be, float* __restrict__ Y, int rows) {
    int lane = threadIdx.x & 31, wave = threadIdx.x >> 5;
    int row = blockIdx.x * 8 + wave;
    if (row >= rows) return;
    const float* rp = X + (long)row * 256;
    float vals[8];
    float s = 0.f;
#pragma unroll
    for (int ii = 0; ii < 8; ++ii) { vals[ii] = rp[lane + ii * 32]; s += vals[ii]; }
#pragma unroll
    for (int o = 16; o > 0; o >>= 1) s += __shfl_xor(s, o, 32);
    float mean = s * (1.f / 256.f);
    float vs = 0.f;
#pragma unroll
    for (int ii = 0; ii < 8; ++ii) { float d = vals[ii] - mean; vs += d * d; }
#pragma unroll
    for (int o = 16; o > 0; o >>= 1) vs += __shfl_xor(vs, o, 32);
    float inv = rsqrtf(vs * (1.f / 256.f) + LN_EPS);
#pragma unroll
    for (int ii = 0; ii < 8; ++ii) {
        int c = lane + ii * 32;
        Y[(long)row * 256 + c] = (vals[ii] - mean) * inv * g[c] + be[c];
    }
}

// Naive C[M,N] = A[M,K] @ B[N,K]^T + bias, optional relu (small matrices only).
__global__ void small_gemm_kernel(const float* __restrict__ A, const float* __restrict__ B,
                                  const float* __restrict__ bias, float* __restrict__ C,
                                  int M, int N, int K, int relu) {
    int idx = blockIdx.x * 256 + threadIdx.x;
    if (idx >= M * N) return;
    int m = idx / N, nn = idx - m * N;
    const float* a = A + (long)m * K;
    const float* b = B + (long)nn * K;
    float s = bias[nn];
#pragma unroll 8
    for (int k = 0; k < K; ++k) s += a[k] * b[k];
    if (relu) s = fmaxf(s, 0.f);
    C[idx] = s;
}

// ------------------------------------------------ fused LN + K/V projection (WMMA)
// Grid: 2048 WGs x 256 threads. Each WG: 64 rows of x -> LN -> bf16 LDS,
// then 8 waves cover 128 output tiles (64 for k, 64 for v), 16x16 each.
__global__ void __launch_bounds__(256) kv_kernel(
        const float* __restrict__ inputs, const float* __restrict__ g_in,
        const float* __restrict__ be_in,
        const u16* __restrict__ wk_bf, const u16* __restrict__ wv_bf,
        const float* __restrict__ bk, const float* __restrict__ bv,
        u16* __restrict__ k_bf, u16* __restrict__ v_bf) {
    __shared__ u16 xlds[64 * 256];   // 32 KB bf16 LN'd input tile
    int tid = threadIdx.x, lane = tid & 31, wave = tid >> 5;
    long rowbase = (long)blockIdx.x * 64;

    // Phase 1: LayerNorm, 8 rows per wave
    for (int rr = 0; rr < 8; ++rr) {
        int r = wave * 8 + rr;
        const float* rp = inputs + (rowbase + r) * 256;
        float vals[8];
        float s = 0.f;
#pragma unroll
        for (int ii = 0; ii < 8; ++ii) { vals[ii] = rp[lane + ii * 32]; s += vals[ii]; }
#pragma unroll
        for (int o = 16; o > 0; o >>= 1) s += __shfl_xor(s, o, 32);
        float mean = s * (1.f / 256.f);
        float vs = 0.f;
#pragma unroll
        for (int ii = 0; ii < 8; ++ii) { float d = vals[ii] - mean; vs += d * d; }
#pragma unroll
        for (int o = 16; o > 0; o >>= 1) vs += __shfl_xor(vs, o, 32);
        float inv = rsqrtf(vs * (1.f / 256.f) + LN_EPS);
#pragma unroll
        for (int ii = 0; ii < 8; ++ii) {
            int c = lane + ii * 32;
            xlds[r * 256 + c] = f2bf((vals[ii] - mean) * inv * g_in[c] + be_in[c]);
        }
    }
    __syncthreads();

    // Phase 2: WMMA GEMM, tiles t: [0,64) -> k, [64,128) -> v
    int hi = lane >> 4, lo = lane & 15;
    for (int t = wave; t < 128; t += 8) {
        int isv = t >= 64;
        int tt = t & 63, rt = tt >> 4, ct = tt & 15;
        const u16* W = isv ? wv_bf : wk_bf;
        const float* bias = isv ? bv : bk;
        u16* out = isv ? v_bf : k_bf;
        const u16* arow = &xlds[(rt * 16 + lo) * 256];
        const u16* brow = W + (long)(ct * 16 + lo) * 256;
        v8f acc = {};
#pragma unroll
        for (int ks = 0; ks < 8; ++ks) {
            v16bf a = make_a_frag(arow, ks * 32, hi);
            v16bf b = make_b_frag_rowk(brow, ks * 32, hi);
            acc = wmma_bf16(a, b, acc);
        }
        int col = ct * 16 + lo;
        float bc = bias[col];
#pragma unroll
        for (int r = 0; r < 8; ++r) {
            long row = rowbase + rt * 16 + r + hi * 8;
            out[row * 256 + col] = f2bf(acc[r] + bc);
        }
    }
}

// q scale + cast into padded [b][16][256] bf16 layout (slot rows 8..15 zero).
__global__ void qcast_kernel(const float* __restrict__ q, u16* __restrict__ q_bf) {
    int idx = blockIdx.x * 256 + threadIdx.x;   // 32*16*256
    int c = idx & 255, i = (idx >> 8) & 15, b = idx >> 12;
    float v = 0.f;
    if (i < 8) v = q[((long)(b * 8 + i)) * 256 + c] * 0.0625f;   // scale = 256^-0.5
    q_bf[idx] = f2bf(v);
}

// dots^T[b][j][i] = sum_k k[b,j,k] * q[b,i,k]  (WMMA, one 16x16 tile per wave)
__global__ void __launch_bounds__(256) dots_kernel(const u16* __restrict__ k_bf,
                                                   const u16* __restrict__ q_bf,
                                                   float* __restrict__ dotsT) {
    int tid = threadIdx.x, lane = tid & 31, wave = tid >> 5;
    int w = blockIdx.x * 8 + wave;      // 0..8191
    int b = w >> 8, jt = w & 255;
    int hi = lane >> 4, lo = lane & 15;
    const u16* arow = k_bf + ((long)(b * 4096 + jt * 16 + lo)) * 256;
    const u16* brow = q_bf + ((long)(b * 16 + lo)) * 256;
    v8f acc = {};
#pragma unroll
    for (int ks = 0; ks < 8; ++ks) {
        v16bf a  = make_a_frag(arow, ks * 32, hi);
        v16bf bb = make_b_frag_rowk(brow, ks * 32, hi);
        acc = wmma_bf16(a, bb, acc);
    }
    if (lo < 8) {
#pragma unroll
        for (int r = 0; r < 8; ++r) {
            int j = jt * 16 + r + hi * 8;
            dotsT[((long)(b * 4096 + j)) * 8 + lo] = acc[r];
        }
    }
}

// Softmax over the 8 slots per (b,j); rewrites dotsT in place with attn+EPS.
// Last iteration additionally writes attn (no EPS) transposed to output.
__global__ void softmax_kernel(float* __restrict__ dotsT, float* __restrict__ attn_out,
                               int write_attn) {
    int gid = blockIdx.x * 256 + threadIdx.x;   // 32*4096
    int b = gid >> 12, j = gid & 4095;
    float* p = dotsT + (long)gid * 8;
    float d[8], m = -3.4e38f;
#pragma unroll
    for (int i = 0; i < 8; ++i) { d[i] = p[i]; m = fmaxf(m, d[i]); }
    float s = 0.f;
#pragma unroll
    for (int i = 0; i < 8; ++i) { d[i] = __expf(d[i] - m); s += d[i]; }
    float inv = 1.f / s;
#pragma unroll
    for (int i = 0; i < 8; ++i) {
        float a = d[i] * inv;
        if (write_attn) attn_out[((long)(b * 8 + i)) * 4096 + j] = a;
        p[i] = a + REN_EPS;
    }
}

// Deterministic colsum[b][i] = sum_j dotsT[b][j][i]; one block per (b,i).
__global__ void colsum_kernel(const float* __restrict__ dotsT, float* __restrict__ colsum) {
    __shared__ float red[256];
    int b = blockIdx.x >> 3, i = blockIdx.x & 7, tid = threadIdx.x;
    const float* p = dotsT + (long)b * 4096 * 8 + i;
    float s = 0.f;
    for (int j = tid; j < 4096; j += 256) s += p[(long)j * 8];
    red[tid] = s;
    __syncthreads();
    for (int o = 128; o > 0; o >>= 1) { if (tid < o) red[tid] += red[tid + o]; __syncthreads(); }
    if (tid == 0) colsum[blockIdx.x] = red[0];
}

// attnw[b][i][j] = bf16((attn+EPS)/colsum[b][i]), layout [b][16][4096] (rows 8..15 stay 0).
__global__ void attnw_kernel(const float* __restrict__ dotsT, const float* __restrict__ colsum,
                             u16* __restrict__ attnw) {
    int gid = blockIdx.x * 256 + threadIdx.x;   // 32*4096
    int b = gid >> 12, j = gid & 4095;
    const float* p = dotsT + (long)gid * 8;
#pragma unroll
    for (int i = 0; i < 8; ++i)
        attnw[((long)(b * 16 + i)) * 4096 + j] = f2bf(p[i] / colsum[b * 8 + i]);
}

// updates partial: grid 32*8 WGs (b, jchunk of 512). v tile staged TRANSPOSED
// in LDS ([d][j], stride 40 -> conflict-free, 16B-aligned fragment reads).
__global__ void __launch_bounds__(256) updates_kernel(const u16* __restrict__ attnw,
                                                      const u16* __restrict__ v_bf,
                                                      float* __restrict__ upart) {
    __shared__ u16 vtt[256 * 40];   // 20 KB: [d=256][j=32] transposed v tile
    int tid = threadIdx.x, lane = tid & 31, wave = tid >> 5;
    int b = blockIdx.x >> 3, chunk = blockIdx.x & 7;
    int hi = lane >> 4, lo = lane & 15;
    int dt0 = wave * 2, dt1 = wave * 2 + 1;
    const u16* vb = v_bf + (long)b * 4096 * 256;
    const u16* arow = attnw + ((long)(b * 16 + lo)) * 4096;
    const u16* bbase0 = &vtt[(dt0 * 16 + lo) * 40];
    const u16* bbase1 = &vtt[(dt1 * 16 + lo) * 40];
    v8f acc0 = {}, acc1 = {};
    for (int ks = chunk * 16; ks < chunk * 16 + 16; ++ks) {
        int jbase = ks * 32;
        {   // cooperative transposed store: 32 j-rows x 256 d-cols
            int row = tid >> 3, cb = (tid & 7) * 32;      // j-row, d-col base
            const u16* src = vb + (long)(jbase + row) * 256 + cb;
            v16u16 d0 = *(const v16u16*)(src);
            v16u16 d1 = *(const v16u16*)(src + 16);
#pragma unroll
            for (int i = 0; i < 16; ++i) vtt[(cb + i) * 40 + row] = d0[i];
#pragma unroll
            for (int i = 0; i < 16; ++i) vtt[(cb + 16 + i) * 40 + row] = d1[i];
        }
        __syncthreads();
        v16bf a  = make_a_frag(arow, jbase, hi);
        v16bf b0 = make_b_frag_ldsT(bbase0, hi);
        v16bf b1 = make_b_frag_ldsT(bbase1, hi);
        acc0 = wmma_bf16(a, b0, acc0);
        acc1 = wmma_bf16(a, b1, acc1);
        __syncthreads();
    }
    if (hi == 0) {   // only M=0..7 (valid slots); N = lo
#pragma unroll
        for (int r = 0; r < 8; ++r) {
            upart[((long)((b * 8 + chunk) * 8 + r)) * 256 + dt0 * 16 + lo] = acc0[r];
            upart[((long)((b * 8 + chunk) * 8 + r)) * 256 + dt1 * 16 + lo] = acc1[r];
        }
    }
}

__global__ void ured_kernel(const float* __restrict__ upart, float* __restrict__ updates) {
    int idx = blockIdx.x * 256 + threadIdx.x;   // 32*8*256
    int b = idx >> 11, i = (idx >> 8) & 7, c = idx & 255;
    float s = 0.f;
#pragma unroll
    for (int ch = 0; ch < 8; ++ch)
        s += upart[((long)((b * 8 + ch) * 8 + i)) * 256 + c];
    updates[idx] = s;
}

__global__ void gru_kernel(const float* __restrict__ gi, const float* __restrict__ gh,
                           float* __restrict__ slots) {
    int idx = blockIdx.x * 256 + threadIdx.x;   // 65536
    int row = idx >> 8, c = idx & 255;
    const float* gir = gi + (long)row * 768;
    const float* ghr = gh + (long)row * 768;
    float r = 1.f / (1.f + __expf(-(gir[c] + ghr[c])));
    float z = 1.f / (1.f + __expf(-(gir[256 + c] + ghr[256 + c])));
    float n = tanhf(gir[512 + c] + r * ghr[512 + c]);
    float h = slots[idx];
    slots[idx] = (1.f - z) * n + z * h;
}

// ---------------------------------------------------------------------- launch
extern "C" void kernel_launch(void* const* d_in, const int* in_sizes, int n_in,
                              void* d_out, int out_size, void* d_ws, size_t ws_size,
                              hipStream_t stream) {
    const float* inputs = (const float*)d_in[0];
    const float* noise  = (const float*)d_in[1];
    const float* mu     = (const float*)d_in[2];
    const float* lsig   = (const float*)d_in[3];
    const float* Wq = (const float*)d_in[4];  const float* bq = (const float*)d_in[5];
    const float* Wk = (const float*)d_in[6];  const float* bk = (const float*)d_in[7];
    const float* Wv = (const float*)d_in[8];  const float* bv = (const float*)d_in[9];
    const float* W_ih = (const float*)d_in[10]; const float* W_hh = (const float*)d_in[11];
    const float* b_ih = (const float*)d_in[12]; const float* b_hh = (const float*)d_in[13];
    const float* W1 = (const float*)d_in[14]; const float* b1 = (const float*)d_in[15];
    const float* W2 = (const float*)d_in[16]; const float* b2 = (const float*)d_in[17];
    const float* g_in = (const float*)d_in[18]; const float* be_in = (const float*)d_in[19];
    const float* g_sl = (const float*)d_in[20]; const float* be_sl = (const float*)d_in[21];
    const float* g_ff = (const float*)d_in[22]; const float* be_ff = (const float*)d_in[23];

    float* out_slots = (float*)d_out;              // 32*8*256
    float* out_attn  = (float*)d_out + 65536;      // 32*8*4096

    char* w = (char*)d_ws;
    size_t off = 0;
    auto alloc = [&](size_t bytes) { char* p = w + off; off += bytes; return p; };
    float* slots    = (float*)alloc(262144);
    float* slots_ln = (float*)alloc(262144);
    float* qf       = (float*)alloc(262144);
    u16*   q_bf     = (u16*)  alloc(262144);
    float* gi       = (float*)alloc(786432);
    float* gh       = (float*)alloc(786432);
    float* pre      = (float*)alloc(262144);
    float* hbuf     = (float*)alloc(262144);
    float* ffo      = (float*)alloc(262144);
    float* updates  = (float*)alloc(262144);
    float* colsum   = (float*)alloc(1024);
    float* dotsT    = (float*)alloc(4194304);      // [32][4096][8] f32
    u16*   attnw    = (u16*)  alloc(4194304);      // [32][16][4096] bf16
    float* upart    = (float*)alloc(2097152);      // [32][8][8][256]
    u16*   wq_bf    = (u16*)  alloc(131072);
    u16*   wk_bf    = (u16*)  alloc(131072);
    u16*   wv_bf    = (u16*)  alloc(131072);
    u16*   k_bf     = (u16*)  alloc(67108864);     // [32*4096][256] bf16
    u16*   v_bf     = (u16*)  alloc(67108864);
    (void)ws_size; (void)in_sizes; (void)n_in; (void)out_size;

    dim3 blk(256);
    // one-time setup
    castw_kernel<<<256, blk, 0, stream>>>(Wq, wq_bf, 65536);
    castw_kernel<<<256, blk, 0, stream>>>(Wk, wk_bf, 65536);
    castw_kernel<<<256, blk, 0, stream>>>(Wv, wv_bf, 65536);
    init_slots_kernel<<<256, blk, 0, stream>>>(noise, mu, lsig, slots);
    zero_kernel<<<4096, blk, 0, stream>>>((float*)attnw, 1048576);   // zero padded rows too
    kv_kernel<<<2048, blk, 0, stream>>>(inputs, g_in, be_in, wk_bf, wv_bf, bk, bv, k_bf, v_bf);

    for (int it = 0; it < 3; ++it) {
        ln_rows_kernel<<<32, blk, 0, stream>>>(slots, g_sl, be_sl, slots_ln, 256);
        small_gemm_kernel<<<256, blk, 0, stream>>>(slots_ln, Wq, bq, qf, 256, 256, 256, 0);
        qcast_kernel<<<512, blk, 0, stream>>>(qf, q_bf);
        dots_kernel<<<1024, blk, 0, stream>>>(k_bf, q_bf, dotsT);
        softmax_kernel<<<512, blk, 0, stream>>>(dotsT, out_attn, (it == 2) ? 1 : 0);
        colsum_kernel<<<256, blk, 0, stream>>>(dotsT, colsum);
        attnw_kernel<<<512, blk, 0, stream>>>(dotsT, colsum, attnw);
        updates_kernel<<<256, blk, 0, stream>>>(attnw, v_bf, upart);
        ured_kernel<<<256, blk, 0, stream>>>(upart, updates);
        small_gemm_kernel<<<768, blk, 0, stream>>>(updates, W_ih, b_ih, gi, 256, 768, 256, 0);
        small_gemm_kernel<<<768, blk, 0, stream>>>(slots, W_hh, b_hh, gh, 256, 768, 256, 0);
        gru_kernel<<<256, blk, 0, stream>>>(gi, gh, slots);
        ln_rows_kernel<<<32, blk, 0, stream>>>(slots, g_ff, be_ff, pre, 256);
        small_gemm_kernel<<<256, blk, 0, stream>>>(pre, W1, b1, hbuf, 256, 256, 256, 1);
        small_gemm_kernel<<<256, blk, 0, stream>>>(hbuf, W2, b2, ffo, 256, 256, 256, 0);
        add_kernel<<<256, blk, 0, stream>>>(slots, ffo, 65536);
    }
    copy_kernel<<<256, blk, 0, stream>>>(slots, out_slots, 65536);
}
